// ViT_78554951844337
// MI455X (gfx1250) — matrix operand order repaired and checked by reading, not compiled, not run on hardware
//
#include <hip/hip_runtime.h>
#include <hip/hip_bf16.h>
#include <math.h>

// ---------------------------------------------------------------------------
// CDNA5 (gfx1250, wave32) WMMA ViT forward — bf16 operands, fp32 accumulate,
// async global->LDS staging (ASYNCcnt) with double-buffered LDS.
// ---------------------------------------------------------------------------
typedef __bf16 bf16_t;
typedef bf16_t v16bf __attribute__((ext_vector_type(16)));
typedef float  v8f   __attribute__((ext_vector_type(8)));

union FragU { v16bf v; uint4 q[2]; };
union US8  { uint4 q; unsigned short s[8]; };

static __device__ __forceinline__ unsigned short f2bf(float f) {
  unsigned int u = __builtin_bit_cast(unsigned int, f);
  unsigned int r = u + 0x7FFFu + ((u >> 16) & 1u);   // round-to-nearest-even
  return (unsigned short)(r >> 16);
}

// ---------------------------------------------------------------------------
// Batched GEMM:  C = alpha * A(MxK) * B(KxN) [+bias] [+Res] [GELU]
// A: bf16 row-major. transB=1 -> B is (N x K) row-major (pre-transposed
// weights / attention K^T): tiles staged with GLOBAL_LOAD_ASYNC_TO_LDS_B128.
// transB=0 (attn@V only) -> B is (K x N), lane-transposed into LDS (sync).
// Block tile 128x128, BK=64, 256 threads = 8 waves (2M x 4N), each wave
// 64x32 = 4x2 fragments; 16 v_wmma per tile, double-buffered LDS pipeline.
// ---------------------------------------------------------------------------
struct GemmParams {
  const unsigned short* A; const unsigned short* B;
  const float* bias; const float* Res;
  void* Cout; int outBF;
  int M, N, K, lda, ldb, ldc, transB, act;
  float alpha;
  long long sAb, sAh, sBb, sBh, sCb, sCh;
  int nH;
};

constexpr int LSTR = 72;   // LDS row stride in shorts (144B, 16B-aligned)

__launch_bounds__(256)
__global__ void gemm_wmma_k(GemmParams p) {
  __shared__ __attribute__((aligned(16))) unsigned short As[2][128 * LSTR];
  __shared__ __attribute__((aligned(16))) unsigned short Bs[2][128 * LSTR];

  const int tid = threadIdx.x;
  const int bz  = blockIdx.z;
  const int bb  = bz / p.nH;
  const int hh  = bz % p.nH;
  const unsigned short* A = p.A + (long long)bb * p.sAb + (long long)hh * p.sAh;
  const unsigned short* B = p.B + (long long)bb * p.sBb + (long long)hh * p.sBh;
  const long long coff = (long long)bb * p.sCb + (long long)hh * p.sCh;

  const long rowBase = (long)blockIdx.y * 128;
  const long colBase = (long)blockIdx.x * 128;
  const int lane  = tid & 31;
  const int wave  = tid >> 5;
  const int waveM = wave >> 2;      // 0..1  (64 rows)
  const int waveN = wave & 3;       // 0..3  (32 cols)
  const int lr = lane & 15;
  const int hi = lane >> 4;

  const bool fullM = (rowBase + 128) <= (long)p.M;
  const bool fullN = (colBase + 128) <= (long)p.N;

  // ---- tile staging (issues async loads or sync zero-filled copies) ----
  auto stageA = [&](long kBase, unsigned short* buf) {
    if (fullM) {
      for (int s = tid; s < 1024; s += 256) {        // 8 x b128 per row
        int r = s >> 3, kc = (s & 7) << 3;
        const unsigned short* gp = A + (rowBase + r) * (long)p.lda + kBase + kc;
        unsigned lds = (unsigned)(unsigned long long)&buf[r * LSTR + kc];
        asm volatile("global_load_async_to_lds_b128 %0, %1, off"
                     :: "v"(lds), "v"((unsigned long long)gp) : "memory");
      }
    } else {
      for (int s = tid; s < 1024; s += 256) {
        int r = s >> 3, kc = (s & 7) << 3;
        long gr = rowBase + r;
        uint4 val = make_uint4(0u, 0u, 0u, 0u);
        if (gr < p.M)
          val = *(const uint4*)(A + gr * (long)p.lda + kBase + kc);
        *(uint4*)&buf[r * LSTR + kc] = val;
      }
    }
  };
  auto stageB = [&](long kBase, unsigned short* buf) {
    if (p.transB) {               // B is (N x K): straight b128 copy
      if (fullN) {
        for (int s = tid; s < 1024; s += 256) {
          int r = s >> 3, kc = (s & 7) << 3;
          const unsigned short* gp = B + (colBase + r) * (long)p.ldb + kBase + kc;
          unsigned lds = (unsigned)(unsigned long long)&buf[r * LSTR + kc];
          asm volatile("global_load_async_to_lds_b128 %0, %1, off"
                       :: "v"(lds), "v"((unsigned long long)gp) : "memory");
        }
      } else {
        for (int s = tid; s < 1024; s += 256) {
          int r = s >> 3, kc = (s & 7) << 3;
          long gn = colBase + r;
          uint4 val = make_uint4(0u, 0u, 0u, 0u);
          if (gn < p.N)
            val = *(const uint4*)(B + gn * (long)p.ldb + kBase + kc);
          *(uint4*)&buf[r * LSTR + kc] = val;
        }
      }
    } else {                      // B is (K x N): vector load, transpose store
      for (int s = tid; s < 1024; s += 256) {
        int k = s >> 4, n8 = (s & 15) << 3;
        long gk = kBase + k;
        US8 u; u.q = make_uint4(0u, 0u, 0u, 0u);
        if (gk < p.K && (colBase + n8 + 7) < p.N)
          u.q = *(const uint4*)(B + gk * (long)p.ldb + colBase + n8);
        #pragma unroll
        for (int j = 0; j < 8; ++j) buf[(n8 + j) * LSTR + k] = u.s[j];
      }
    }
  };

  v8f acc[4][2] = {};

  const int ktiles = (p.K + 63) >> 6;
  // prologue: stage tile 0 into buffer 0
  stageA(0, As[0]);
  stageB(0, Bs[0]);

  for (int kt = 0; kt < ktiles; ++kt) {
    const int X = kt & 1;
    // drain this wave's async writes, then rendezvous: after this barrier the
    // current buffer is fully written and nobody still reads the other buffer
    asm volatile("s_wait_asynccnt 0" ::: "memory");
    __syncthreads();
    if (kt + 1 < ktiles) {        // stage next tile under this tile's compute
      const long kNext = (long)(kt + 1) * 64;
      stageA(kNext, As[X ^ 1]);
      stageB(kNext, Bs[X ^ 1]);
    }

    // ---- 2 K-steps of WMMA on buffer X ----
    #pragma unroll
    for (int kk = 0; kk < 2; ++kk) {
      FragU a[4], b[2];
      #pragma unroll
      for (int mi = 0; mi < 4; ++mi) {
        const uint4* q = (const uint4*)&As[X][(waveM * 64 + mi * 16 + lr) * LSTR];
        a[mi].q[0] = q[kk * 4 + hi];        // K = kk*32 + hi*8 + 0..7
        a[mi].q[1] = q[kk * 4 + hi + 2];    // K = kk*32 + 16 + hi*8 + 0..7
      }
      #pragma unroll
      for (int ni = 0; ni < 2; ++ni) {
        const uint4* q = (const uint4*)&Bs[X][(waveN * 32 + ni * 16 + lr) * LSTR];
        b[ni].q[0] = q[kk * 4 + hi];         // K = kk*32 + hi*8 + 0..7
        b[ni].q[1] = q[kk * 4 + hi + 2];     // K = kk*32 + 16 + hi*8 + 0..7
      }
      #pragma unroll
      for (int mi = 0; mi < 4; ++mi)
        #pragma unroll
        for (int ni = 0; ni < 2; ++ni)
          acc[mi][ni] = __builtin_amdgcn_wmma_f32_16x16x32_bf16(
              false, a[mi].v, false, b[ni].v, (short)0, acc[mi][ni], false, false);
    }
  }

  // ---- epilogue ----
  #pragma unroll
  for (int mi = 0; mi < 4; ++mi)
    #pragma unroll
    for (int ni = 0; ni < 2; ++ni)
      #pragma unroll
      for (int v = 0; v < 8; ++v) {
        long gr = rowBase + waveM * 64 + mi * 16 + hi * 8 + v;
        long gc = colBase + waveN * 32 + ni * 16 + lr;
        if (gr < p.M && gc < p.N) {
          float val = acc[mi][ni][v] * p.alpha;
          if (p.bias) val += p.bias[gc];
          if (p.Res)  val += p.Res[gr * (long)p.ldc + gc];
          if (p.act == 1) val = 0.5f * val * (1.0f + erff(val * 0.70710678118f));
          long long o = coff + gr * (long)p.ldc + gc;
          if (p.outBF) ((unsigned short*)p.Cout)[o] = f2bf(val);
          else         ((float*)p.Cout)[o] = val;
        }
      }
}

// ---------------------------------------------------------------------------
// Support kernels
// ---------------------------------------------------------------------------
__global__ void wtransT_k(const float* __restrict__ W, unsigned short* __restrict__ WT,
                          int K, int N) {
  long idx = (long)blockIdx.x * blockDim.x + threadIdx.x;
  long total = (long)K * N;
  if (idx >= total) return;
  int n = (int)(idx % N);
  long k = idx / N;
  WT[(long)n * K + k] = f2bf(W[idx]);
}

__global__ void patchify_k(const float* __restrict__ x, unsigned short* __restrict__ out) {
  long idx = (long)blockIdx.x * blockDim.x + threadIdx.x;
  const long total = 32L * 576 * 768;
  if (idx >= total) return;
  int col = (int)(idx % 768);
  long rr = idx / 768;
  int pj = (int)(rr % 24);
  long r2 = rr / 24;
  int pi = (int)(r2 % 24);
  int b  = (int)(r2 / 24);
  int pw = col & 15, ph = (col >> 4) & 15, c = col >> 8;
  out[idx] = f2bf(x[(((long)b * 3 + c) * 384 + (pi * 16 + ph)) * 384 + (pj * 16 + pw)]);
}

__global__ void assemble_k(const float* __restrict__ zp, const float* __restrict__ cls,
                           const float* __restrict__ pos, float* __restrict__ z) {
  long idx = (long)blockIdx.x * blockDim.x + threadIdx.x;
  const long total = 32L * 577 * 768;
  if (idx >= total) return;
  int l = (int)(idx % 768);
  long rt = idx / 768;
  int t = (int)(rt % 577);
  int b = (int)(rt / 577);
  float v = pos[(long)b * 768 + l];
  v += (t == 0) ? cls[(long)b * 768 + l]
                : zp[((long)b * 576 + (t - 1)) * 768 + l];
  z[idx] = v;
}

__launch_bounds__(256)
__global__ void layernorm_k(const float* __restrict__ in, unsigned short* __restrict__ out,
                            const float* __restrict__ g, const float* __restrict__ b,
                            int C, long long inStride, long long outStride) {
  __shared__ float sm[256];
  const float* ip = in + (long long)blockIdx.x * inStride;
  unsigned short* op = out + (long long)blockIdx.x * outStride;
  int tid = threadIdx.x;

  float s = 0.0f;
  for (int i = tid; i < C; i += 256) s += ip[i];
  sm[tid] = s; __syncthreads();
  for (int st = 128; st > 0; st >>= 1) { if (tid < st) sm[tid] += sm[tid + st]; __syncthreads(); }
  float mean = sm[0] / (float)C; __syncthreads();

  float vs = 0.0f;
  for (int i = tid; i < C; i += 256) { float d = ip[i] - mean; vs += d * d; }
  sm[tid] = vs; __syncthreads();
  for (int st = 128; st > 0; st >>= 1) { if (tid < st) sm[tid] += sm[tid + st]; __syncthreads(); }
  float rstd = rsqrtf(sm[0] / (float)C + 1e-5f); __syncthreads();

  for (int i = tid; i < C; i += 256)
    op[i] = f2bf((ip[i] - mean) * rstd * g[i] + b[i]);
}

__launch_bounds__(256)
__global__ void softmaxbf_k(const float* __restrict__ S, unsigned short* __restrict__ O,
                            int n, int np) {
  __shared__ float sm[256];
  const float* row = S + (long)blockIdx.x * n;
  unsigned short* orow = O + (long)blockIdx.x * np;
  int tid = threadIdx.x;

  float m = -3.4e38f;
  for (int i = tid; i < n; i += 256) m = fmaxf(m, row[i]);
  sm[tid] = m; __syncthreads();
  for (int st = 128; st > 0; st >>= 1) { if (tid < st) sm[tid] = fmaxf(sm[tid], sm[tid + st]); __syncthreads(); }
  m = sm[0]; __syncthreads();

  float s = 0.0f;
  for (int i = tid; i < n; i += 256) s += __expf(row[i] - m);
  sm[tid] = s; __syncthreads();
  for (int st = 128; st > 0; st >>= 1) { if (tid < st) sm[tid] += sm[tid + st]; __syncthreads(); }
  float inv = 1.0f / sm[0]; __syncthreads();

  for (int i = tid; i < np; i += 256) {
    float e = (i < n) ? __expf(row[i] - m) * inv : 0.0f;
    orow[i] = f2bf(e);
  }
}

// ---------------------------------------------------------------------------
// Host orchestration
// ---------------------------------------------------------------------------
static inline void launch_gemm(hipStream_t stream,
    const unsigned short* A, const unsigned short* B,
    const float* bias, const float* Res, void* Cout, int outBF,
    int M, int N, int K, int lda, int ldb, int ldc, int transB, int act, float alpha,
    int nbatch = 1, int nH = 1,
    long long sAb = 0, long long sAh = 0, long long sBb = 0, long long sBh = 0,
    long long sCb = 0, long long sCh = 0) {
  GemmParams p{A, B, bias, Res, Cout, outBF, M, N, K, lda, ldb, ldc, transB, act,
               alpha, sAb, sAh, sBb, sBh, sCb, sCh, nH};
  dim3 grid((unsigned)((N + 127) / 128), (unsigned)((M + 127) / 128), (unsigned)nbatch);
  gemm_wmma_k<<<grid, dim3(256), 0, stream>>>(p);
}

extern "C" void kernel_launch(void* const* d_in, const int* in_sizes, int n_in,
                              void* d_out, int out_size, void* d_ws, size_t ws_size,
                              hipStream_t stream) {
  (void)in_sizes; (void)n_in; (void)out_size; (void)ws_size;
  const float* x    = (const float*)d_in[0];
  const float* Wp   = (const float*)d_in[1];
  const float* bp   = (const float*)d_in[2];
  const float* cls  = (const float*)d_in[3];
  const float* pos  = (const float*)d_in[4];
  const float* ln_g = (const float*)d_in[5];
  const float* ln_b = (const float*)d_in[6];
  const float* Wqkv = (const float*)d_in[7];
  const float* Wo   = (const float*)d_in[8];
  const float* bo   = (const float*)d_in[9];
  const float* W1   = (const float*)d_in[10];
  const float* b1   = (const float*)d_in[11];
  const float* W2   = (const float*)d_in[12];
  const float* b2   = (const float*)d_in[13];
  const float* hg   = (const float*)d_in[14];
  const float* hb   = (const float*)d_in[15];
  const float* hW1  = (const float*)d_in[16];
  const float* hb1  = (const float*)d_in[17];
  const float* hW2  = (const float*)d_in[18];
  const float* hb2  = (const float*)d_in[19];

  constexpr int Bn = 32, NPAT = 576, NTOK = 577, LD = 768, NH = 12, HD = 64;
  constexpr int D3 = 2304, DF = 3072, NC = 1000, DEPTH = 12, CB = 4;
  constexpr int SBP = 640;                  // padded softmax width (= 10*64)
  constexpr long TOK   = (long)Bn * NTOK;   // 18464
  constexpr long PROWS = (long)Bn * NPAT;   // 18432

  char* base = (char*)d_ws;
  size_t off = 0;
  auto takeB = [&](size_t bytes) -> void* {
    off = (off + 255) & ~(size_t)255;
    void* p = base + off;
    off += bytes;
    return p;
  };
  typedef unsigned short u16;
  u16*   patches = (u16*)  takeB((size_t)PROWS * LD * 2);
  float* zp      = (float*)takeB((size_t)PROWS * LD * 4);
  float* z       = (float*)takeB((size_t)TOK * LD * 4);
  float* z1      = (float*)takeB((size_t)TOK * LD * 4);
  u16*   hn      = (u16*)  takeB((size_t)TOK * LD * 2);
  u16*   qkvb    = (u16*)  takeB((size_t)TOK * D3 * 2);
  u16*   obuf    = (u16*)  takeB((size_t)TOK * LD * 2);
  u16*   mlpb    = (u16*)  takeB((size_t)TOK * DF * 2);
  float* scores  = (float*)takeB((size_t)CB * NH * NTOK * NTOK * 4);
  u16*   sbf     = (u16*)  takeB((size_t)CB * NH * NTOK * SBP * 2);
  u16*   ctb     = (u16*)  takeB((size_t)Bn * LD * 2);
  u16*   h1b     = (u16*)  takeB((size_t)Bn * LD * 2);
  u16* WpT   = (u16*)takeB((size_t)LD * LD * 2);
  u16* WqkvT = (u16*)takeB((size_t)DEPTH * D3 * LD * 2);
  u16* WoT   = (u16*)takeB((size_t)DEPTH * LD * LD * 2);
  u16* W1T   = (u16*)takeB((size_t)DEPTH * DF * LD * 2);
  u16* W2T   = (u16*)takeB((size_t)DEPTH * LD * DF * 2);
  u16* hW1T  = (u16*)takeB((size_t)LD * LD * 2);
  u16* hW2T  = (u16*)takeB((size_t)NC * LD * 2);

  auto wt = [&](const float* W, u16* WT, int K, int N) {
    long tot = (long)K * N;
    wtransT_k<<<(unsigned)((tot + 255) / 256), 256, 0, stream>>>(W, WT, K, N);
  };
  wt(Wp, WpT, LD, LD);
  for (int i = 0; i < DEPTH; ++i) {
    wt(Wqkv + (size_t)i * LD * D3, WqkvT + (size_t)i * D3 * LD, LD, D3);
    wt(Wo   + (size_t)i * LD * LD, WoT   + (size_t)i * LD * LD, LD, LD);
    wt(W1   + (size_t)i * LD * DF, W1T   + (size_t)i * DF * LD, LD, DF);
    wt(W2   + (size_t)i * DF * LD, W2T   + (size_t)i * LD * DF, DF, LD);
  }
  wt(hW1, hW1T, LD, LD);
  wt(hW2, hW2T, LD, NC);

  // 1) patch embed
  {
    long total = PROWS * LD;
    patchify_k<<<(unsigned)((total + 255) / 256), 256, 0, stream>>>(x, patches);
    launch_gemm(stream, patches, WpT, bp, nullptr, zp, /*outBF=*/0,
                (int)PROWS, LD, LD, LD, LD, LD, /*transB=*/1, 0, 1.0f);
    long ztot = TOK * LD;
    assemble_k<<<(unsigned)((ztot + 255) / 256), 256, 0, stream>>>(zp, cls, pos, z);
  }

  const long long S2 = (long long)NTOK * NTOK;
  const long long SP = (long long)NTOK * SBP;

  // 2) transformer blocks
  for (int i = 0; i < DEPTH; ++i) {
    const float* g  = ln_g + (size_t)i * LD;
    const float* bl = ln_b + (size_t)i * LD;

    layernorm_k<<<(unsigned)TOK, 256, 0, stream>>>(z, hn, g, bl, LD, LD, LD);
    launch_gemm(stream, hn, WqkvT + (size_t)i * D3 * LD, nullptr, nullptr,
                qkvb, /*outBF=*/1,
                (int)TOK, D3, LD, LD, LD, D3, /*transB=*/1, 0, 1.0f);

    for (int cb = 0; cb < Bn; cb += CB) {
      const u16* qa = qkvb + (size_t)cb * NTOK * D3;
      // scores = (1/8) q @ k^T  [48 x 577 x 577] fp32
      launch_gemm(stream, qa, qa + LD, nullptr, nullptr, scores, /*outBF=*/0,
                  NTOK, NTOK, HD, D3, D3, NTOK, /*transB=*/1, 0, 0.125f,
                  CB * NH, NH,
                  (long long)NTOK * D3, HD,
                  (long long)NTOK * D3, HD,
                  (long long)NH * S2, S2);
      softmaxbf_k<<<(unsigned)(CB * NH * NTOK), 256, 0, stream>>>(scores, sbf, NTOK, SBP);
      // o = softmax @ v -> obuf [B,577,768] bf16
      launch_gemm(stream, sbf, qa + 2 * LD, nullptr, nullptr,
                  obuf + (size_t)cb * NTOK * LD, /*outBF=*/1,
                  NTOK, HD, NTOK, SBP, D3, LD, /*transB=*/0, 0, 1.0f,
                  CB * NH, NH,
                  (long long)NH * SP, SP,
                  (long long)NTOK * D3, HD,
                  (long long)NTOK * LD, HD);
    }

    launch_gemm(stream, obuf, WoT + (size_t)i * LD * LD, bo + (size_t)i * LD, z,
                z1, /*outBF=*/0,
                (int)TOK, LD, LD, LD, LD, LD, /*transB=*/1, 0, 1.0f);
    layernorm_k<<<(unsigned)TOK, 256, 0, stream>>>(z1, hn, g, bl, LD, LD, LD);
    launch_gemm(stream, hn, W1T + (size_t)i * DF * LD, b1 + (size_t)i * DF, nullptr,
                mlpb, /*outBF=*/1,
                (int)TOK, DF, LD, LD, LD, DF, /*transB=*/1, /*gelu=*/1, 1.0f);
    launch_gemm(stream, mlpb, W2T + (size_t)i * LD * DF, b2 + (size_t)i * LD, z1,
                z, /*outBF=*/0,
                (int)TOK, LD, DF, DF, DF, LD, /*transB=*/1, 0, 1.0f);
  }

  // 3) head
  layernorm_k<<<(unsigned)Bn, 256, 0, stream>>>(z, ctb, hg, hb, LD,
                                                (long long)NTOK * LD, LD);
  launch_gemm(stream, ctb, hW1T, hb1, nullptr, h1b, /*outBF=*/1,
              Bn, LD, LD, LD, LD, LD, /*transB=*/1, 0, 1.0f);
  launch_gemm(stream, h1b, hW2T, hb2, nullptr, (float*)d_out, /*outBF=*/0,
              Bn, NC, LD, LD, LD, NC, /*transB=*/1, 0, 1.0f);
}
